// GATNet_53240414601405
// MI455X (gfx1250) — compile-verified, matrix operand order checked by reading, not run on hardware
//
#include <hip/hip_runtime.h>
#include <hip/hip_bf16.h>
#include <math.h>

typedef __attribute__((ext_vector_type(16))) _Float16 v16h;
typedef __attribute__((ext_vector_type(8)))  float    v8f;

#define NN    20480
#define EDGES 81920
#define ETOT  102400   // EDGES + NN self loops
#define BATCH 512

static __device__ __forceinline__ void edge_sd(const int* __restrict__ ei, int e, int& s, int& d) {
    if (e < EDGES) { s = ei[e]; d = ei[EDGES + e]; }
    else           { s = e - EDGES; d = s; }
}

static __device__ __forceinline__ void atomicMaxF(float* addr, float val) {
    int* ia = (int*)addr;
    int cur = __float_as_int(*addr);
    while (val > __int_as_float(cur)) {
        int prev = atomicCAS(ia, cur, __float_as_int(val));
        if (prev == cur) break;
        cur = prev;
    }
}

// ---------------------------------------------------------------------------
// Weight pre-passes: make the B operand fragment-addressable exactly like A.
// Bt[n*K + k] = (f16) B[k*N + n]   -> lane's 8-element K-runs are contiguous,
// so B fragments become global_load_b128 of f16, no cvt in the GEMM loop.
// ---------------------------------------------------------------------------
__global__ void transpose_f16(const float* __restrict__ B, _Float16* __restrict__ Bt,
                              int K, int N)
{
    int t = blockIdx.x * blockDim.x + threadIdx.x;
    if (t >= K * N) return;
    int n = t / K, k = t - n * K;
    Bt[t] = (_Float16)B[k * N + n];
}

__global__ void tof16(const float* __restrict__ s, _Float16* __restrict__ d, int n)
{
    int t = blockIdx.x * blockDim.x + threadIdx.x;
    if (t < n) d[t] = (_Float16)s[t];
}

// ---------------------------------------------------------------------------
// Generic GEMM via v_wmma_f32_16x16x32_f16 (f32 accumulate).
// A: f32 [M,K] row-major;  Bt: f16 [N,K] row-major (pre-transposed weights).
// C[row*ldc + coloff + col] = act(A @ B + bias[col]);  ACT: 0 none, 1 relu
// One wave computes one 16x16 tile; K-loop in steps of 32.
// ---------------------------------------------------------------------------
template<int ACT>
__global__ void wmma_gemm(const float* __restrict__ A, const _Float16* __restrict__ Bt,
                          const float* __restrict__ bias, float* __restrict__ C,
                          int M, int N, int K, int ldc, int coloff)
{
    const int wavesPerBlock = blockDim.x >> 5;
    const int wid  = blockIdx.x * wavesPerBlock + (threadIdx.x >> 5);
    const int lane = threadIdx.x & 31;
    const int tilesM = (M + 15) >> 4, tilesN = (N + 15) >> 4;
    if (wid >= tilesM * tilesN) return;             // wave-uniform exit
    const int tileM = wid / tilesN, tileN = wid % tilesN;
    const int r  = lane & 15;
    const int hi = lane >> 4;
    const int rowA = tileM * 16 + r;                // A row held by this lane
    const int colB = tileN * 16 + r;                // B column held by this lane

    v8f acc = {};
    for (int k0 = 0; k0 < K; k0 += 32) {
        if (rowA < M && k0 + 32 < K)
            __builtin_prefetch(&A[rowA * K + k0 + 32], 0, 1);
        v16h a, b;
#pragma unroll
        for (int e = 0; e < 16; ++e) {
            const int kk = k0 + ((e < 8) ? 0 : 16) + hi * 8 + (e & 7);
            const float    av = (rowA < M && kk < K) ? A[rowA * K + kk] : 0.0f;
            const _Float16 bv = (colB < N && kk < K) ? Bt[colB * K + kk] : (_Float16)0.0f;
            a[e] = (_Float16)av;
            b[e] = bv;
        }
        acc = __builtin_amdgcn_wmma_f32_16x16x32_f16(false, a, false, b,
                                                     (short)0, acc, false, false);
    }
#pragma unroll
    for (int i = 0; i < 8; ++i) {
        const int row = tileM * 16 + i + hi * 8;
        const int col = tileN * 16 + r;
        if (row < M && col < N) {
            float v = acc[i];
            if (bias) v += bias[col];
            if (ACT == 1) v = fmaxf(v, 0.0f);
            C[row * ldc + coloff + col] = v;
        }
    }
}

// ---------------------------------------------------------------------------
// Protein branch conv1d as WMMA GEMM with LDS-resident f16 embedding table.
// out[b,o,h] = relu( sum_{c,k} emb[target[b,c], h+k] * convW[o,c,k] + convB[o] )
// M = 512*121 rows (b,h), N = 32, K = 8000.  convW16 is the f16 cast of conv_w
// (already [N,K]=Bt layout).  Stored flattened as convact[b*3872 + o*121 + h]
// so fc_xt1 reads a plain [512,3872] matrix.
// ---------------------------------------------------------------------------
__global__ void conv_wmma(const float* __restrict__ emb, const int* __restrict__ target,
                          const _Float16* __restrict__ convW16,
                          const float* __restrict__ convB, float* __restrict__ convact)
{
    __shared__ _Float16 semb[26 * 128];             // 6.5 KB of the 320 KB WGP LDS
    for (int i = threadIdx.x; i < 26 * 128; i += blockDim.x)
        semb[i] = (_Float16)emb[i];
    __syncthreads();

    const int TILESN = 2;                           // N=32 -> 2 tiles
    const int TM = (BATCH * 121) / 16;              // 3872 M-tiles (exact)
    const int wavesPerBlock = blockDim.x >> 5;
    const int wid  = blockIdx.x * wavesPerBlock + (threadIdx.x >> 5);
    const int lane = threadIdx.x & 31;
    const int tileM = wid / TILESN, tileN = wid % TILESN;
    if (tileM >= TM) return;
    const int r  = lane & 15;
    const int hi = lane >> 4;

    const int rowA  = tileM * 16 + r;
    const int b     = rowA / 121;
    const int hh    = rowA % 121;
    const int tbase = b * 1000;
    const int o     = tileN * 16 + r;               // 0..31
    const _Float16* wrow = convW16 + o * 8000;

    v8f acc = {};
    for (int k0 = 0; k0 < 8000; k0 += 32) {
        // lane's 16 A elements split into two runs of 8 consecutive K -> only
        // two target lookups, then LDS f16 gathers from the embedding table.
        const int cbase = (k0 + hi * 8) >> 3;
        const int t0 = target[tbase + cbase];
        const int t1 = target[tbase + cbase + 2];
        const _Float16* p0 = &semb[t0 * 128 + hh];
        const _Float16* p1 = &semb[t1 * 128 + hh];
        v16h a, bm;
#pragma unroll
        for (int e = 0; e < 8; ++e) { a[e] = p0[e]; a[8 + e] = p1[e]; }
#pragma unroll
        for (int e = 0; e < 16; ++e) {
            const int kk = k0 + ((e < 8) ? 0 : 16) + hi * 8 + (e & 7);
            bm[e] = wrow[kk];
        }
        acc = __builtin_amdgcn_wmma_f32_16x16x32_f16(false, a, false, bm,
                                                     (short)0, acc, false, false);
    }
    const float bo = convB[o];
#pragma unroll
    for (int i = 0; i < 8; ++i) {
        const int m  = tileM * 16 + i + hi * 8;
        const int bb = m / 121, h2 = m % 121;
        convact[bb * 3872 + o * 121 + h2] = fmaxf(acc[i] + bo, 0.0f);
    }
}

// ---------------------------------------------------------------------------
// GAT glue kernels
// ---------------------------------------------------------------------------
__global__ void fillk(float* p, int n, float v) {
    int t = blockIdx.x * blockDim.x + threadIdx.x;
    if (t < n) p[t] = v;
}

// per-node, per-head attention scores: ss = <xl[n,h,:], a_src[h,:]>, sd likewise
__global__ void gat_scores(const float* __restrict__ xl, const float* __restrict__ asrc,
                           const float* __restrict__ adst, int H, int C,
                           float* __restrict__ ss, float* __restrict__ sd)
{
    int t = blockIdx.x * blockDim.x + threadIdx.x;
    if (t >= NN * H) return;
    int n = t / H, h = t - n * H;
    const float* xp = xl + (size_t)n * H * C + h * C;
    const float* ap = asrc + h * C;
    const float* bp = adst + h * C;
    float s0 = 0.f, s1 = 0.f;
    for (int c = 0; c < C; ++c) { float xv = xp[c]; s0 += xv * ap[c]; s1 += xv * bp[c]; }
    ss[t] = s0; sd[t] = s1;
}

__global__ void gat_pass1(const int* __restrict__ ei, const float* __restrict__ ss,
                          const float* __restrict__ sd, int H, float* __restrict__ m)
{
    int t = blockIdx.x * blockDim.x + threadIdx.x;
    if (t >= ETOT * H) return;
    int e = t / H, h = t - e * H;
    int s, d; edge_sd(ei, e, s, d);
    float v = ss[s * H + h] + sd[d * H + h];
    v = (v >= 0.f) ? v : 0.2f * v;                  // leaky_relu(0.2)
    atomicMaxF(&m[d * H + h], v);
}

__global__ void gat_pass2(const int* __restrict__ ei, const float* __restrict__ ss,
                          const float* __restrict__ sd, const float* __restrict__ m,
                          int H, float* __restrict__ z)
{
    int t = blockIdx.x * blockDim.x + threadIdx.x;
    if (t >= ETOT * H) return;
    int e = t / H, h = t - e * H;
    int s, d; edge_sd(ei, e, s, d);
    float v = ss[s * H + h] + sd[d * H + h];
    v = (v >= 0.f) ? v : 0.2f * v;
    atomicAdd(&z[d * H + h], __expf(v - m[d * H + h]));
}

__global__ void gat_pass3(const int* __restrict__ ei, const float* __restrict__ ss,
                          const float* __restrict__ sd, const float* __restrict__ m,
                          const float* __restrict__ z, const float* __restrict__ xl,
                          int H, int C, float* __restrict__ out)
{
    int t = blockIdx.x * blockDim.x + threadIdx.x;
    if (t >= ETOT * H) return;
    int e = t / H, h = t - e * H;
    int s, d; edge_sd(ei, e, s, d);
    float v = ss[s * H + h] + sd[d * H + h];
    v = (v >= 0.f) ? v : 0.2f * v;
    float alpha = __expf(v - m[d * H + h]) / z[d * H + h];
    const float* xp = xl + (size_t)s * H * C + h * C;
    float*       op = out + (size_t)d * H * C + h * C;
    for (int c = 0; c < C; ++c) atomicAdd(&op[c], alpha * xp[c]);
}

// out[i] = act(out[i] + bias[i % F]);  ACT: 1 relu, 2 elu
template<int ACT>
__global__ void bias_act(float* __restrict__ buf, const float* __restrict__ bias,
                         int total, int F)
{
    int t = blockIdx.x * blockDim.x + threadIdx.x;
    if (t >= total) return;
    float v = buf[t] + bias[t % F];
    if (ACT == 1) v = fmaxf(v, 0.f);
    else          v = (v > 0.f) ? v : (__expf(v) - 1.f);
    buf[t] = v;
}

// graphs are 40 consecutive nodes (batch = repeat(arange(512), 40))
__global__ void maxpool40(const float* __restrict__ h, float* __restrict__ g)
{
    int t = blockIdx.x * blockDim.x + threadIdx.x;
    if (t >= BATCH * 128) return;
    int b = t >> 7, c = t & 127;
    const float* p = h + (size_t)b * 40 * 128 + c;
    float m = p[0];
    for (int i = 1; i < 40; ++i) m = fmaxf(m, p[i * 128]);
    g[t] = m;
}

// ---------------------------------------------------------------------------
static inline int cdiv(int a, int b) { return (a + b - 1) / b; }

extern "C" void kernel_launch(void* const* d_in, const int* in_sizes, int n_in,
                              void* d_out, int out_size, void* d_ws, size_t ws_size,
                              hipStream_t stream)
{
    const float* x        = (const float*)d_in[0];
    const int*   ei       = (const int*)  d_in[1];
    // d_in[2] = batch: structure (40 consecutive nodes/graph) exploited directly
    const int*   target   = (const int*)  d_in[3];
    const float* W1       = (const float*)d_in[4];
    const float* a_src1   = (const float*)d_in[5];
    const float* a_dst1   = (const float*)d_in[6];
    const float* b1       = (const float*)d_in[7];
    const float* W2       = (const float*)d_in[8];
    const float* a_src2   = (const float*)d_in[9];
    const float* a_dst2   = (const float*)d_in[10];
    const float* b2       = (const float*)d_in[11];
    const float* fc_g1_w  = (const float*)d_in[12];
    const float* fc_g1_b  = (const float*)d_in[13];
    const float* emb      = (const float*)d_in[14];
    const float* conv_w   = (const float*)d_in[15];
    const float* conv_b   = (const float*)d_in[16];
    const float* fc_xt1_w = (const float*)d_in[17];
    const float* fc_xt1_b = (const float*)d_in[18];
    const float* fc1_w    = (const float*)d_in[19];
    const float* fc1_b    = (const float*)d_in[20];
    const float* fc2_w    = (const float*)d_in[21];
    const float* fc2_b    = (const float*)d_in[22];
    const float* out_w    = (const float*)d_in[23];
    const float* out_b    = (const float*)d_in[24];
    float* out = (float*)d_out;
    float* ws  = (float*)d_ws;

    // ---- workspace layout (floats), arenas reused across phases ----
    const size_t SZ_XL1 = (size_t)NN * 780;          // 15,974,400
    float* xl1  = ws;                                // arena 1 (64 MB)
    float* h1   = ws + SZ_XL1;                       // arena 2 (64 MB)
    float* misc = ws + 2 * SZ_XL1;
    float* ss1 = misc;            float* sd1 = misc + 204800;
    float* m1  = misc + 409600;   float* z1  = misc + 614400;
    // phase B (reuses arena 1 + misc)
    float* xl2 = xl1;                                // 20480*128
    float* h2  = xl1 + 3000000;                      // 20480*128
    float* ss2 = misc;            float* sd2 = misc + 20480;
    float* m2  = misc + 40960;    float* z2  = misc + 61440;
    // phase C (arena 1 tail + arena 2)
    float* g       = xl1 + 6000000;                  // 512*128
    float* xc      = xl1 + 6100000;                  // 512*256 fused concat
    float* t1      = xl1 + 6300000;                  // 512*1024
    float* t2      = xl1 + 6900000;                  // 512*256
    float* convact = h1;                             // 512*3872 (h1 dead by then)
    // shared f16 weight strip (max user = fc_xt1_w: 3872*128 = 495,616 halves)
    _Float16* Bt   = (_Float16*)(ws + 32800000);

    const int TB = 256;                              // 8 waves / block
    auto gemmGrid = [](int M, int N) { return cdiv(cdiv(M,16) * cdiv(N,16), 8); };
    auto prepGrid = [](int n) { return cdiv(n, 256); };

    // ================= GAT layer 1 (78 -> 10 heads x 78, ELU) =================
    fillk<<<cdiv((int)SZ_XL1, TB), TB, 0, stream>>>(h1, (int)SZ_XL1, 0.f);
    fillk<<<cdiv(204800, TB), TB, 0, stream>>>(m1, 204800, -3.0e38f);
    fillk<<<cdiv(204800, TB), TB, 0, stream>>>(z1, 204800, 0.f);
    transpose_f16<<<prepGrid(78*780), TB, 0, stream>>>(W1, Bt, 78, 780);
    wmma_gemm<0><<<gemmGrid(NN,780), TB, 0, stream>>>(x, Bt, nullptr, xl1, NN, 780, 78, 780, 0);
    gat_scores<<<cdiv(NN*10, TB), TB, 0, stream>>>(xl1, a_src1, a_dst1, 10, 78, ss1, sd1);
    gat_pass1<<<cdiv(ETOT*10, TB), TB, 0, stream>>>(ei, ss1, sd1, 10, m1);
    gat_pass2<<<cdiv(ETOT*10, TB), TB, 0, stream>>>(ei, ss1, sd1, m1, 10, z1);
    gat_pass3<<<cdiv(ETOT*10, TB), TB, 0, stream>>>(ei, ss1, sd1, m1, z1, xl1, 10, 78, h1);
    bias_act<2><<<cdiv((int)SZ_XL1, TB), TB, 0, stream>>>(h1, b1, (int)SZ_XL1, 780);

    // ================= GAT layer 2 (780 -> 128, 1 head, ReLU) =================
    fillk<<<cdiv(NN*128, TB), TB, 0, stream>>>(h2, NN*128, 0.f);
    fillk<<<cdiv(20480, TB), TB, 0, stream>>>(m2, 20480, -3.0e38f);
    fillk<<<cdiv(20480, TB), TB, 0, stream>>>(z2, 20480, 0.f);
    transpose_f16<<<prepGrid(780*128), TB, 0, stream>>>(W2, Bt, 780, 128);
    wmma_gemm<0><<<gemmGrid(NN,128), TB, 0, stream>>>(h1, Bt, nullptr, xl2, NN, 128, 780, 128, 0);
    gat_scores<<<cdiv(NN, TB), TB, 0, stream>>>(xl2, a_src2, a_dst2, 1, 128, ss2, sd2);
    gat_pass1<<<cdiv(ETOT, TB), TB, 0, stream>>>(ei, ss2, sd2, 1, m2);
    gat_pass2<<<cdiv(ETOT, TB), TB, 0, stream>>>(ei, ss2, sd2, m2, 1, z2);
    gat_pass3<<<cdiv(ETOT, TB), TB, 0, stream>>>(ei, ss2, sd2, m2, z2, xl2, 1, 128, h2);
    bias_act<1><<<cdiv(NN*128, TB), TB, 0, stream>>>(h2, b2, NN*128, 128);

    // ============ pool + protein conv + fused concat ============
    maxpool40<<<cdiv(BATCH*128, TB), TB, 0, stream>>>(h2, g);
    tof16<<<prepGrid(32*8000), TB, 0, stream>>>(conv_w, Bt, 32*8000);
    conv_wmma<<<cdiv(3872*2, 8), TB, 0, stream>>>(emb, target, Bt, conv_b, convact);
    transpose_f16<<<prepGrid(128*128), TB, 0, stream>>>(fc_g1_w, Bt, 128, 128);
    wmma_gemm<1><<<gemmGrid(BATCH,128), TB, 0, stream>>>(g, Bt, fc_g1_b, xc,
                                                         BATCH, 128, 128, 256, 0);
    transpose_f16<<<prepGrid(3872*128), TB, 0, stream>>>(fc_xt1_w, Bt, 3872, 128);
    wmma_gemm<0><<<gemmGrid(BATCH,128), TB, 0, stream>>>(convact, Bt, fc_xt1_b, xc,
                                                         BATCH, 128, 3872, 256, 128);

    // ============ fusion MLP ============
    transpose_f16<<<prepGrid(256*1024), TB, 0, stream>>>(fc1_w, Bt, 256, 1024);
    wmma_gemm<1><<<gemmGrid(BATCH,1024), TB, 0, stream>>>(xc, Bt, fc1_b, t1,
                                                          BATCH, 1024, 256, 1024, 0);
    transpose_f16<<<prepGrid(1024*256), TB, 0, stream>>>(fc2_w, Bt, 1024, 256);
    wmma_gemm<1><<<gemmGrid(BATCH,256), TB, 0, stream>>>(t1, Bt, fc2_b, t2,
                                                         BATCH, 256, 1024, 256, 0);
    transpose_f16<<<prepGrid(256*1), TB, 0, stream>>>(out_w, Bt, 256, 1);
    wmma_gemm<0><<<gemmGrid(BATCH,1), TB, 0, stream>>>(t2, Bt, out_b, out,
                                                       BATCH, 1, 256, 1, 0);
}
